// InClusterisationLoss_21930103013689
// MI455X (gfx1250) — matrix operand order repaired and checked by reading, not compiled, not run on hardware
//
#include <hip/hip_runtime.h>
#include <hip/hip_bf16.h>

#define EPS 1e-6f

typedef __attribute__((ext_vector_type(16))) __bf16        v16bf;
typedef __attribute__((ext_vector_type(8)))  float         v8f;
typedef __attribute__((ext_vector_type(8)))  unsigned int  v8u;
typedef __attribute__((ext_vector_type(4)))  int           v4i;

#if defined(__HIP_DEVICE_COMPILE__) && __has_builtin(__builtin_amdgcn_global_load_async_to_lds_b128)
#define HAVE_ASYNC_LDS 1
#else
#define HAVE_ASYNC_LDS 0
#endif

#define GLOBAL_AS __attribute__((address_space(1)))
#define LDS_AS    __attribute__((address_space(3)))

// ---- fp32 -> bf16 with round-to-nearest-even ----
__device__ __forceinline__ unsigned int f2bf1(float f) {
    unsigned int x = __float_as_uint(f);
    return (x + 0x7fffu + ((x >> 16) & 1u)) >> 16;
}
__device__ __forceinline__ unsigned int f2bf2(float lo, float hi) {
    return f2bf1(lo) | (f2bf1(hi) << 16);
}

// ------------------------------------------------------------------
// 1) Pre-swizzle centroids into WMMA A-fragment layout (bf16).
//    Fragment f = kt*16 + dk covers centroids [kt*16, kt*16+16) and
//    d-range [dk*32, dk*32+32).  A 16x32 k order used here (contiguous):
//      lanes 0-15 : row M=lane, K = {0..15}
//      lanes16-31 : row M=lane-16, K = {16..31}
//    Storage: packedA[(f*32 + lane)*16 + h], h = halfword 0..15.
// ------------------------------------------------------------------
__global__ void pack_centroids(const float* __restrict__ cent,
                               unsigned short* __restrict__ packedA,
                               int K) {
    int e = blockIdx.x * blockDim.x + threadIdx.x;
    if (e >= K * 512) return;
    int h  = e & 15;
    int l  = (e >> 4) & 31;
    int f  = e >> 9;
    int kt = f >> 4;
    int dk = f & 15;
    int m  = l & 15;
    int kk;
    if (l < 16) kk = h;                             // K in {0..15}
    else        kk = h + 16;                        // K in {16..31}
    int d = dk * 32 + kk;
    float v = cent[(size_t)(kt * 16 + m) * 512 + d];
    packedA[e] = (unsigned short)f2bf1(v);
}

// ------------------------------------------------------------------
// 2) Per-centroid constant c[k] = ||a||^2 + 2*eps*sum(a)  (one wave/centroid)
// ------------------------------------------------------------------
__global__ void centroid_consts(const float* __restrict__ cent,
                                float* __restrict__ cvals, int K) {
    int wave = (blockIdx.x * blockDim.x + threadIdx.x) >> 5;
    int lane = threadIdx.x & 31;
    if (wave >= K) return;
    const float* row = cent + (size_t)wave * 512;
    float n = 0.f, s = 0.f;
    for (int d = lane; d < 512; d += 32) {
        float x = row[d];
        n += x * x;
        s += x;
    }
    for (int off = 16; off; off >>= 1) {
        n += __shfl_xor(n, off, 32);
        s += __shfl_xor(s, off, 32);
    }
    if (lane == 0) cvals[wave] = n + 2.0f * EPS * s;
}

// ------------------------------------------------------------------
// 3) Zero the per-centroid accumulators (required every launch).
// ------------------------------------------------------------------
__global__ void init_accum(float* __restrict__ sums, float* __restrict__ counts,
                           unsigned int* __restrict__ maxb, int K) {
    int i = blockIdx.x * blockDim.x + threadIdx.x;
    if (i < K) { sums[i] = 0.f; counts[i] = 0.f; maxb[i] = 0u; }
}

// ------------------------------------------------------------------
// Stage one 16 KB centroid tile (16 A-fragments) into LDS.
// 256 threads x 4 x 16B.  Async path: ASYNCcnt-tracked DMA into LDS
// (4 instructions per wave per tile, no VGPR round trip).
// ------------------------------------------------------------------
__device__ __forceinline__ void stage_tile(const unsigned short* __restrict__ g,
                                           unsigned short* l, int tid) {
#if HAVE_ASYNC_LDS
    GLOBAL_AS v4i* gp = (GLOBAL_AS v4i*)(g + (size_t)tid * 8);  // 16 B / thread
    LDS_AS   v4i* lp = (LDS_AS   v4i*)(l + (size_t)tid * 8);
#pragma unroll
    for (int i = 0; i < 4; ++i)                                  // 4 KB strides
        __builtin_amdgcn_global_load_async_to_lds_b128(gp + i * 256,
                                                       lp + i * 256, 0, 0);
#else
    const uint4* gs = (const uint4*)g;
    uint4*       ls = (uint4*)l;
#pragma unroll
    for (int i = 0; i < 4; ++i) ls[tid + i * 256] = gs[tid + i * 256];
#endif
}

// ------------------------------------------------------------------
// 4) Hot kernel: each wave owns 16 embeddings (B frags in VGPRs);
//    the block's 8 waves share centroid A-tiles staged in LDS
//    (double-buffered async DMA) -> 8x less L2 traffic.
// ------------------------------------------------------------------
__global__ __launch_bounds__(256) void
assign_accumulate(const float* __restrict__ emb,
                  const unsigned short* __restrict__ packedA,
                  const float* __restrict__ cvals,
                  float* __restrict__ sums,
                  float* __restrict__ counts,
                  unsigned int* __restrict__ maxb,
                  int K) {
    __shared__ float cl[1024];
    __shared__ __align__(16) unsigned short abuf[2][8192];  // 2 x 16 KB tiles

    const int tid = threadIdx.x;
    for (int i = tid; i < K; i += 256) cl[i] = cvals[i];

    const int  lane = tid & 31;
    const int  wid  = tid >> 5;
    const int  col  = lane & 15;   // embedding within the wave's 16-tile
    const int  half = lane >> 4;   // which 16-d sub-chunk this lane packs
    const long nbase = ((long)blockIdx.x * 8 + wid) * 16;
    const float* rowp = emb + (nbase + col) * 512 + half * 16;

    const int KT = K >> 4;
    // Prefetch tiles 0 and 1 while we pack B fragments.
    stage_tile(packedA, abuf[0], tid);
    if (KT > 1) stage_tile(packedA + 8192, abuf[1], tid);

    // --- load the wave's 16 embeddings once; exact fp32 norms; pack B frags.
    //     B 32x16 bf16 k order used here (same as A): lanes 0-15 hold K=0..15 (col N=lane),
    //     lanes 16-31 hold K=16..31 - i.e. 16 consecutive d per lane. ---
    v16bf bfrag[16];
    float nrm = 0.f, ssum = 0.f;
#pragma unroll
    for (int dk = 0; dk < 16; ++dk) {
        const float4* p = reinterpret_cast<const float4*>(rowp + dk * 32);
        float4 x0 = p[0], x1 = p[1], x2 = p[2], x3 = p[3];
        nrm  += x0.x*x0.x + x0.y*x0.y + x0.z*x0.z + x0.w*x0.w
              + x1.x*x1.x + x1.y*x1.y + x1.z*x1.z + x1.w*x1.w
              + x2.x*x2.x + x2.y*x2.y + x2.z*x2.z + x2.w*x2.w
              + x3.x*x3.x + x3.y*x3.y + x3.z*x3.z + x3.w*x3.w;
        ssum += x0.x + x0.y + x0.z + x0.w + x1.x + x1.y + x1.z + x1.w
              + x2.x + x2.y + x2.z + x2.w + x3.x + x3.y + x3.z + x3.w;
        v8u pk;
        pk[0] = f2bf2(x0.x, x0.y); pk[1] = f2bf2(x0.z, x0.w);
        pk[2] = f2bf2(x1.x, x1.y); pk[3] = f2bf2(x1.z, x1.w);
        pk[4] = f2bf2(x2.x, x2.y); pk[5] = f2bf2(x2.z, x2.w);
        pk[6] = f2bf2(x3.x, x3.y); pk[7] = f2bf2(x3.z, x3.w);
        bfrag[dk] = __builtin_bit_cast(v16bf, pk);
    }
    // lane L and L+16 together cover the full 512-d row of embedding col.
    float nrmO = __shfl_xor(nrm, 16, 32);
    float sO   = __shfl_xor(ssum, 16, 32);
    const float bc = (nrm + nrmO) - 2.0f * EPS * (ssum + sO)
                   + 512.0f * EPS * EPS;   // ||b||^2 - 2*eps*sum(b) + d*eps^2

    float bestV = 3.0e38f;
    int   bestI = 0;
    const int mofs = half << 3;            // C/D layout: M = r + 8*(lane>=16)

    for (int kt = 0; kt < KT; ++kt) {
#if HAVE_ASYNC_LDS
        // one tile (4 async ops/wave) may remain in flight; current tile done
        if (kt < KT - 1) __builtin_amdgcn_s_wait_asynccnt(4);
        else             __builtin_amdgcn_s_wait_asynccnt(0);
#endif
        __syncthreads();                       // tile (kt&1) visible to all waves

        const unsigned short* atile = &abuf[kt & 1][lane * 16];
        v8f acc = {0.f, 0.f, 0.f, 0.f, 0.f, 0.f, 0.f, 0.f};

        // Software-pipelined A-fragment reads: prefetch distance 2 so each
        // WMMA overlaps with the two ds_load_b128 of fragment dk+2.
        v16bf a0 = *reinterpret_cast<const v16bf*>(atile);
        v16bf a1 = *reinterpret_cast<const v16bf*>(atile + 512);
#pragma unroll
        for (int dk = 0; dk < 16; ++dk) {
            v16bf an = a0;
            if (dk + 2 < 16)
                an = *reinterpret_cast<const v16bf*>(atile + (dk + 2) * 512);
            acc = __builtin_amdgcn_wmma_f32_16x16x32_bf16(
                false, a0, false, bfrag[dk], (short)0, acc, false, false);
            a0 = a1;
            a1 = an;
        }
        const float* cp = &cl[kt * 16 + mofs];
#pragma unroll
        for (int r = 0; r < 8; ++r) {
            float score = cp[r] - 2.0f * acc[r];   // c[k] - 2*<a,b>
            if (score < bestV) { bestV = score; bestI = kt * 16 + mofs + r; }
        }
        __syncthreads();                       // everyone done reading tile (kt&1)
        if (kt + 2 < KT)
            stage_tile(packedA + (size_t)(kt + 2) * 8192, abuf[kt & 1], tid);
    }

    // merge the two M-halves; ties -> lower centroid index (jnp.argmin).
    float vO = __shfl_xor(bestV, 16, 32);
    int   iO = __shfl_xor(bestI, 16, 32);
    if (vO < bestV || (vO == bestV && iO < bestI)) { bestV = vO; bestI = iO; }

    if (lane < 16) {
        float dist = sqrtf(fmaxf(bestV + bc, 0.0f));
        atomicAdd(&sums[bestI], dist);
        atomicAdd(&counts[bestI], 1.0f);
        atomicMax(&maxb[bestI], __float_as_uint(dist));  // dist>=0: bit max == fp max
    }
}

// ------------------------------------------------------------------
// 5) Deterministic final reduction to the 3 output scalars.
// ------------------------------------------------------------------
__global__ void finalize(const float* __restrict__ sums,
                         const float* __restrict__ counts,
                         const unsigned int* __restrict__ maxb,
                         float* __restrict__ out, int K) {
    __shared__ float red0[256], red1[256], red2[256];
    int t = threadIdx.x;
    float s0 = 0.f, s1 = 0.f, s2 = 0.f;
    for (int k = t; k < K; k += 256) {
        float c = counts[k];
        s0 += sums[k] / (c + 1.0f);
        s1 += __uint_as_float(maxb[k]);   // already >= 0 (0-init)
        s2 += c;
    }
    red0[t] = s0; red1[t] = s1; red2[t] = s2;
    __syncthreads();
    for (int off = 128; off > 0; off >>= 1) {
        if (t < off) {
            red0[t] += red0[t + off];
            red1[t] += red1[t + off];
            red2[t] += red2[t + off];
        }
        __syncthreads();
    }
    if (t == 0) {
        float invK = 1.0f / (float)K;
        out[0] = red0[0] * invK;   // in_cluster_dist
        out[1] = red1[0] * invK;   // centroids_radius
        out[2] = red2[0] * invK;   // centroids_elements_count
    }
}

extern "C" void kernel_launch(void* const* d_in, const int* in_sizes, int n_in,
                              void* d_out, int out_size, void* d_ws, size_t ws_size,
                              hipStream_t stream) {
    const float* emb  = (const float*)d_in[0];   // (N, 512) fp32
    const float* cent = (const float*)d_in[1];   // (K, 512) fp32
    const int D = 512;
    const int N = in_sizes[0] / D;               // 65536
    const int K = in_sizes[1] / D;               // 1024

    char* ws = (char*)d_ws;
    unsigned short* packedA = (unsigned short*)ws;                  // K*512 u16 (1 MB)
    size_t off = (size_t)K * 512 * sizeof(unsigned short);
    float* cvals  = (float*)(ws + off); off += (size_t)K * sizeof(float);
    float* sums   = (float*)(ws + off); off += (size_t)K * sizeof(float);
    float* cnts   = (float*)(ws + off); off += (size_t)K * sizeof(float);
    unsigned int* maxb = (unsigned int*)(ws + off);

    pack_centroids  <<<(K * 512 + 255) / 256, 256, 0, stream>>>(cent, packedA, K);
    centroid_consts <<<(K * 32 + 255) / 256,  256, 0, stream>>>(cent, cvals, K);
    init_accum      <<<(K + 255) / 256,       256, 0, stream>>>(sums, cnts, maxb, K);
    // 8 waves/block, 16 embeddings/wave -> 128 embeddings per block
    assign_accumulate<<<N / 128, 256, 0, stream>>>(emb, packedA, cvals,
                                                   sums, cnts, maxb, K);
    finalize        <<<1, 256, 0, stream>>>(sums, cnts, maxb, (float*)d_out, K);
}
